// LinearAttention_57475252355201
// MI455X (gfx1250) — matrix operand order repaired and checked by reading, not compiled, not run on hardware
//
#include <hip/hip_runtime.h>
#include <hip/hip_bf16.h>
#include <math.h>

// ---------------------------------------------------------------------------
// XCA linear attention for MI455X (gfx1250, wave32, WMMA bf16 16x16x32).
// Pipeline (all intermediates bf16, fp32 accumulation inside WMMA):
//   1) rmsnorm_bf16_kernel     : x -> xn (l2norm * sqrt(d) * gamma), bf16
//   2) transpose_bf16_kernel   : w_qkv, w_out -> [N][K] bf16 (B-fragment friendly)
//   3) gemm_qkv_kernel         : qkv = xn @ w_qkv, scattered to [3][b*h][dh][n]
//   4) xca_attn_kernel         : row norms + S=QK^T (WMMA) + softmax + attn@V
//   5) gemm_out_kernel         : out = attn_out @ w_out (fp32 result)
// ---------------------------------------------------------------------------

typedef __bf16 bf16;
typedef __attribute__((ext_vector_type(16))) __bf16 v16bf;
typedef __attribute__((ext_vector_type(8)))  __bf16 v8bf;
typedef __attribute__((ext_vector_type(4)))  __bf16 v4bf;
typedef __attribute__((ext_vector_type(8)))  float  v8f;

#define DIM    1024
#define HEADS  16
#define DH     64
#define NTOK   4096
#define BATCH  8
#define M_TOT  (BATCH * NTOK)     // 32768 tokens
#define N_QKV  (3 * HEADS * DH)   // 3072
#define BHN    (BATCH * HEADS)    // 128

// --------------------------- 1) RMSNorm -> bf16 ----------------------------
__global__ void rmsnorm_bf16_kernel(const float* __restrict__ x,
                                    const float* __restrict__ gamma,
                                    bf16* __restrict__ xn) {
  const int token = blockIdx.x;
  const int tid   = threadIdx.x;  // 256 threads, 4 floats each
  const float4 xv = ((const float4*)(x + (size_t)token * DIM))[tid];
  float ss = xv.x * xv.x + xv.y * xv.y + xv.z * xv.z + xv.w * xv.w;
#pragma unroll
  for (int o = 16; o > 0; o >>= 1) ss += __shfl_xor(ss, o, 32);
  __shared__ float wsum[8];
  if ((tid & 31) == 0) wsum[tid >> 5] = ss;
  __syncthreads();
  float tot = 0.f;
#pragma unroll
  for (int i = 0; i < 8; ++i) tot += wsum[i];
  // x / max(||x||, 1e-12) * sqrt(1024) * gamma
  const float inv = 32.0f / fmaxf(sqrtf(tot), 1e-12f);
  const float4 gv = ((const float4*)gamma)[tid];
  v4bf o;
  o.x = (__bf16)(xv.x * inv * gv.x);
  o.y = (__bf16)(xv.y * inv * gv.y);
  o.z = (__bf16)(xv.z * inv * gv.z);
  o.w = (__bf16)(xv.w * inv * gv.w);
  ((v4bf*)(xn + (size_t)token * DIM))[tid] = o;
}

// ------------------- 2) weight transpose + bf16 convert --------------------
// w: [K][N] fp32 row-major  ->  wT: [N][K] bf16 (contiguous K per output row)
__global__ void transpose_bf16_kernel(const float* __restrict__ w,
                                      bf16* __restrict__ wT, int K, int N) {
  long idx = (long)blockIdx.x * blockDim.x + threadIdx.x;
  if (idx >= (long)K * N) return;
  int n = (int)(idx / K);
  int k = (int)(idx - (long)n * K);
  wT[idx] = (__bf16)w[(size_t)k * N + n];
}

// A-fragment loader: 16x32 bf16 A. Lane (lm = lane%16 is M-row, half = lane/16)
// holds K = [half*16, half*16+16) in elems 0..15.
__device__ __forceinline__ v16bf load_a_frag(const bf16* __restrict__ row,
                                             int k0, int half) {
  v16bf a = *(const v16bf*)(row + k0 + half * 16);
  return a;
}

// ------------------------- 3) QKV GEMM (WMMA bf16) -------------------------
// C[32768][3072] = xn @ w_qkv ; each wave owns a 16x64 strip (4 wmma tiles,
// A fragment reused 4x). Epilogue scatters into qkv[3][b*16+h][dh][n] bf16:
// the WMMA C layout gives each lane 8 consecutive tokens -> one b128 store.
__global__ void gemm_qkv_kernel(const bf16* __restrict__ A,    // [32768][1024]
                                const bf16* __restrict__ BT,   // [3072][1024]
                                bf16* __restrict__ qkv) {      // [3][128][64][4096]
  const int K = DIM;
  const int wave = (int)(((size_t)blockIdx.x * blockDim.x + threadIdx.x) >> 5);
  const int lane = threadIdx.x & 31;
  const int half = lane >> 4, lm = lane & 15;
  const int nStrips = N_QKV / 64;              // 48
  const int m0 = (wave / nStrips) * 16;
  const int n0 = (wave % nStrips) * 64;
  const bf16* arow = A + (size_t)(m0 + lm) * K;
  v8f c[4] = {};
  for (int k0 = 0; k0 < K; k0 += 32) {
    v16bf a = load_a_frag(arow, k0, half);
#pragma unroll
    for (int s = 0; s < 4; ++s) {
      const v16bf b = *(const v16bf*)(BT + (size_t)(n0 + s * 16 + lm) * K +
                                      k0 + half * 16);
      c[s] = __builtin_amdgcn_wmma_f32_16x16x32_bf16(false, a, false, b,
                                                     (short)0, c[s], false, false);
    }
  }
  const int m  = m0 + half * 8;          // first of 8 consecutive tokens
  const int bb = m >> 12;                // batch index (4096 tokens per b)
  const int t  = m & (NTOK - 1);
#pragma unroll
  for (int s = 0; s < 4; ++s) {
    const int j   = n0 + s * 16 + lm;    // column in [0,3072)
    const int qi  = j >> 10;             // 0=q, 1=k, 2=v
    const int rem = j & 1023;
    const int h   = rem >> 6, d = rem & 63;
    bf16* dst = qkv + (((size_t)qi * BHN + bb * HEADS + h) * DH + d) * NTOK + t;
    v8bf o;
#pragma unroll
    for (int r = 0; r < 8; ++r) o[r] = (__bf16)c[s][r];
    *(v8bf*)dst = o;                     // 8 consecutive n, 16B aligned
  }
}

// ------------------- 4) XCA attention per (b,h): 128 blocks ----------------
__global__ void xca_attn_kernel(const bf16* __restrict__ qkv,
                                const float* __restrict__ temperature,
                                bf16* __restrict__ out) {   // [32768][1024]
  const int bh   = blockIdx.x;          // b*16 + h
  const int h    = bh & (HEADS - 1);
  const int tid  = threadIdx.x;         // 256 = 8 waves
  const int wave = tid >> 5, lane = tid & 31;
  const int half = lane >> 4, lm = lane & 15;
  const bf16* q = qkv + (size_t)bh * DH * NTOK;
  const bf16* k = q + (size_t)BHN * DH * NTOK;
  const bf16* v = k + (size_t)BHN * DH * NTOK;

  __shared__ float inv_nrm[2 * DH];     // 64 q norms then 64 k norms
  __shared__ float s_mat[DH][DH + 1];   // scores / attention (padded)

  // Phase 1: 1/max(||row||,eps) over n for 64 q rows + 64 k rows (16 per wave)
#pragma unroll 1
  for (int rr = 0; rr < 16; ++rr) {
    const int row = wave * 16 + rr;
    const bf16* src = (row < DH) ? (q + (size_t)row * NTOK)
                                 : (k + (size_t)(row - DH) * NTOK);
    float ss = 0.f;
    for (int i = lane * 8; i < NTOK; i += 32 * 8) {
      v8bf vv = *(const v8bf*)(src + i);
#pragma unroll
      for (int u = 0; u < 8; ++u) { float f = (float)vv[u]; ss += f * f; }
    }
#pragma unroll
    for (int o = 16; o > 0; o >>= 1) ss += __shfl_xor(ss, o, 32);
    if (lane == 0) inv_nrm[row] = 1.0f / fmaxf(sqrtf(ss), 1e-12f);
  }
  __syncthreads();

  // Phase 2: S = Q K^T over n (WMMA, K-loop = 4096), scaled into LDS.
  const float tsc = 8.0f * expf(temperature[h]);   // ATTN_SCALE * exp(T_h)
#pragma unroll
  for (int p = 0; p < 2; ++p) {
    const int tile = wave + p * 8;                 // 16 tiles of S (4x4)
    const int d0 = (tile >> 2) * 16, e0 = (tile & 3) * 16;
    const bf16* ar = q + (size_t)(d0 + lm) * NTOK;
    const bf16* br = k + (size_t)(e0 + lm) * NTOK;
    v8f c = {};
    for (int k0 = 0; k0 < NTOK; k0 += 32) {
      v16bf a = load_a_frag(ar, k0, half);
      const v16bf b = *(const v16bf*)(br + k0 + half * 16);
      c = __builtin_amdgcn_wmma_f32_16x16x32_bf16(false, a, false, b,
                                                  (short)0, c, false, false);
    }
    const int e = e0 + lm;
#pragma unroll
    for (int r = 0; r < 8; ++r) {
      const int d = d0 + half * 8 + r;
      s_mat[d][e] = c[r] * tsc * inv_nrm[d] * inv_nrm[DH + e];
    }
  }
  __syncthreads();

  // Phase 3: softmax over e (64-wide rows), one thread per row.
  if (tid < DH) {
    float mx = -3.4e38f;
#pragma unroll
    for (int e = 0; e < DH; ++e) mx = fmaxf(mx, s_mat[tid][e]);
    float sum = 0.f;
#pragma unroll
    for (int e = 0; e < DH; ++e) {
      float ex = expf(s_mat[tid][e] - mx);
      s_mat[tid][e] = ex;
      sum += ex;
    }
    const float is = 1.0f / sum;
#pragma unroll
    for (int e = 0; e < DH; ++e) s_mat[tid][e] *= is;
  }
  __syncthreads();

  // Phase 4: out[d][n] = sum_e attn[d][e] * v[e][n]; each thread owns one
  // n-column (coalesced v loads across lanes, LDS broadcast of attn), writes
  // its 64 outputs contiguously in the [token][h*64+d] layout for GEMM2.
  const size_t token_base = (size_t)(bh >> 4) * NTOK;   // b * 4096
  for (int n = tid; n < NTOK; n += 256) {
    float acc[DH];
#pragma unroll
    for (int d = 0; d < DH; ++d) acc[d] = 0.f;
    for (int e = 0; e < DH; ++e) {
      const float vv = (float)v[(size_t)e * NTOK + n];
#pragma unroll
      for (int d = 0; d < DH; ++d) acc[d] += s_mat[d][e] * vv;
    }
    bf16* dst = out + (token_base + n) * DIM + h * DH;
#pragma unroll
    for (int d0 = 0; d0 < DH; d0 += 8) {
      v8bf o;
#pragma unroll
      for (int r = 0; r < 8; ++r) o[r] = (__bf16)acc[d0 + r];
      *(v8bf*)(dst + d0) = o;
    }
  }
}

// ----------------------- 5) output GEMM (WMMA bf16) ------------------------
__global__ void gemm_out_kernel(const bf16* __restrict__ A,    // [32768][1024]
                                const bf16* __restrict__ BT,   // [1024][1024]
                                float* __restrict__ C) {       // [32768][1024]
  const int K = DIM, N = DIM;
  const int wave = (int)(((size_t)blockIdx.x * blockDim.x + threadIdx.x) >> 5);
  const int lane = threadIdx.x & 31;
  const int half = lane >> 4, lm = lane & 15;
  const int nStrips = N / 64;                   // 16
  const int m0 = (wave / nStrips) * 16;
  const int n0 = (wave % nStrips) * 64;
  const bf16* arow = A + (size_t)(m0 + lm) * K;
  v8f c[4] = {};
  for (int k0 = 0; k0 < K; k0 += 32) {
    v16bf a = load_a_frag(arow, k0, half);
#pragma unroll
    for (int s = 0; s < 4; ++s) {
      const v16bf b = *(const v16bf*)(BT + (size_t)(n0 + s * 16 + lm) * K +
                                      k0 + half * 16);
      c[s] = __builtin_amdgcn_wmma_f32_16x16x32_bf16(false, a, false, b,
                                                     (short)0, c[s], false, false);
    }
  }
#pragma unroll
  for (int s = 0; s < 4; ++s) {
    const int j = n0 + s * 16 + lm;
#pragma unroll
    for (int r = 0; r < 8; ++r)
      C[(size_t)(m0 + half * 8 + r) * N + j] = c[s][r];
  }
}

// ---------------------------------------------------------------------------
extern "C" void kernel_launch(void* const* d_in, const int* in_sizes, int n_in,
                              void* d_out, int out_size, void* d_ws, size_t ws_size,
                              hipStream_t stream) {
  const float* x           = (const float*)d_in[0];  // [8,4096,1024]
  const float* gamma       = (const float*)d_in[1];  // [1024]
  const float* w_qkv       = (const float*)d_in[2];  // [1024,3072]
  const float* temperature = (const float*)d_in[3];  // [16,1,1]
  const float* w_out       = (const float*)d_in[4];  // [1024,1024]
  float* outp = (float*)d_out;                       // [8,4096,1024]

  // Workspace layout (328 MB total):
  //   xn     bf16 [32768][1024]           @ 0        (64 MB)
  //   wqkvT  bf16 [3072][1024]            @ 64 MB    ( 6 MB)
  //   woutT  bf16 [1024][1024]            @ 70 MB    ( 2 MB)
  //   qkv    bf16 [3][128][64][4096]      @ 72 MB    (192 MB)
  //   aout   bf16 [32768][1024]           @ 264 MB   (64 MB)
  char* ws = (char*)d_ws;
  bf16* xn    = (bf16*)(ws);
  bf16* wqkvT = (bf16*)(ws + ((size_t)64 << 20));
  bf16* woutT = (bf16*)(ws + ((size_t)70 << 20));
  bf16* qkv   = (bf16*)(ws + ((size_t)72 << 20));
  bf16* aout  = (bf16*)(ws + ((size_t)264 << 20));

  rmsnorm_bf16_kernel<<<M_TOT, 256, 0, stream>>>(x, gamma, xn);
  transpose_bf16_kernel<<<(N_QKV * DIM) / 256, 256, 0, stream>>>(w_qkv, wqkvT,
                                                                 DIM, N_QKV);
  transpose_bf16_kernel<<<(DIM * DIM) / 256, 256, 0, stream>>>(w_out, woutT,
                                                               DIM, DIM);
  // waves = (M/16)*(N/64); 8 waves per 256-thread block
  gemm_qkv_kernel<<<(M_TOT / 16) * (N_QKV / 64) / 8, 256, 0, stream>>>(xn, wqkvT,
                                                                       qkv);
  xca_attn_kernel<<<BHN, 256, 0, stream>>>(qkv, temperature, aout);
  gemm_out_kernel<<<(M_TOT / 16) * (DIM / 64) / 8, 256, 0, stream>>>(aout, woutT,
                                                                     outp);
}
